// GraphTransformer_670014898297
// MI455X (gfx1250) — matrix operand order repaired and checked by reading, not compiled, not run on hardware
//
#include <hip/hip_runtime.h>
#include <hip/hip_bf16.h>

// ---------------------------------------------------------------------------
// GraphTransformer forward for MI455X (gfx1250, wave32, WMMA).
// All GEMMs + attention on v_wmma_f32_16x16x32_f16 (f16 in, f32 accum).
// All hot-loop operands staged so each fragment is 2x global_load_b128:
//   - weights pre-transposed to f16 (NxK)
//   - V stored transposed (per (b,h,d) row over padded tokens)
//   - activations dual-stored f32 (residual) + f16 (GEMM A operand)
// Fragment loads batched in PAIRS (16 transient VGPRs) to avoid spilling;
// adj gather is branchless (clamped loads + cndmask selects), inlined.
// Softmax/LN/GELU epilogues in f32. Workspace use ~92 MB.
// ---------------------------------------------------------------------------

typedef __attribute__((ext_vector_type(16))) _Float16 v16h;
typedef __attribute__((ext_vector_type(8)))  _Float16 v8h;
typedef __attribute__((ext_vector_type(8)))  float    v8f;
typedef __attribute__((ext_vector_type(4)))  float    f32x4;
typedef __attribute__((ext_vector_type(4)))  unsigned int u32x4;

#define BATCH   32
#define SEQ     513            // 512 nodes + cls
#define NTOK    (BATCH*SEQ)    // 16416 == 1026 * 16
#define NHEAD   8
#define HDIM    64
#define QKW     1024           // Q|K halves per token in QK buffer
#define SEQP    544            // padded tokens for V^T rows (17*32)
#define NKEYBLK 17

__device__ __forceinline__ float gelu_exact(float x) {
  return 0.5f * x * (1.0f + erff(x * 0.70710678118654752440f));
}
__device__ __forceinline__ int imin(int a, int b) { return a < b ? a : b; }
__device__ __forceinline__ int imax(int a, int b) { return a > b ? a : b; }

// two contiguous 8-half chunks -> v16h fragment
__device__ __forceinline__ v16h ldfrag_h(const _Float16* p0, const _Float16* p1) {
  union { v16h h; u32x4 u[2]; } x;
  x.u[0] = *(const u32x4*)p0;
  x.u[1] = *(const u32x4*)p1;
  return x.h;
}
// two contiguous 8-float chunks -> v16h fragment (convert)
__device__ __forceinline__ v16h ldfrag_f(const float* p0, const float* p1) {
  f32x4 a0 = *(const f32x4*)p0, a1 = *(const f32x4*)(p0 + 4);
  f32x4 b0 = *(const f32x4*)p1, b1 = *(const f32x4*)(p1 + 4);
  v16h h;
#pragma unroll
  for (int j = 0; j < 4; ++j) {
    h[j]      = (_Float16)a0[j];
    h[4 + j]  = (_Float16)a1[j];
    h[8 + j]  = (_Float16)b0[j];
    h[12 + j] = (_Float16)b1[j];
  }
  return h;
}

// ---------------------------------------------------------------------------
// Generic WMMA GEMM: Out(MxN) = act(A(MxK) @ B(KxN) + bias + res)
// Bt is the f16 *transposed* weight (N x K). One wave owns NT 16x16 tiles
// along N (A fragment reused). M mult of 16, K mult of 32, N guarded.
// OUTM: 0 = f32 row-major, 1 = f16 row-major, 2 = f16 V^T layout
//       Vt[( (row/SEQ)*512 + ncol ) * SEQP + row%SEQ]
// ---------------------------------------------------------------------------
template <bool A_HALF, int ACT, bool HAS_RES, int OUTM, int NT>
__global__ void gemm16(const void* __restrict__ Ap, const void* __restrict__ Btp,
                       const float* __restrict__ bias, const float* __restrict__ res,
                       void* __restrict__ Outp, int M, int N, int K) {
  const int wave = threadIdx.x >> 5;
  const int lane = threadIdx.x & 31;
  const int hf   = lane >> 4;
  const int ln16 = lane & 15;
  const int n0   = (blockIdx.x * 8 + wave) * (NT * 16);
  if (n0 >= N) return;                        // wave-uniform
  const int m0   = blockIdx.y * 16;
  const int mrow = m0 + ln16;

  const float*    Af = (const float*)Ap;
  const _Float16* Ah = (const _Float16*)Ap;
  const _Float16* Bt = (const _Float16*)Btp;

  v8f zero = {};
  v8f acc[NT];
#pragma unroll
  for (int t = 0; t < NT; ++t) acc[t] = zero;

  for (int k0 = 0; k0 < K; k0 += 32) {
    v16h a;
    if (A_HALF) {
      const _Float16* ab = Ah + (size_t)mrow * K + k0;
      a = ldfrag_h(ab + hf * 8, ab + 16 + hf * 8);
    } else {
      const float* ab = Af + (size_t)mrow * K + k0;
      a = ldfrag_f(ab + hf * 8, ab + 16 + hf * 8);
    }
    // B fragments in pairs: 2 loads in flight per 2 WMMAs (low pressure)
#pragma unroll
    for (int t2 = 0; t2 < NT; t2 += 2) {
      v16h bfr[2];
#pragma unroll
      for (int j = 0; j < 2 && t2 + j < NT; ++j) {
        const int ncol = n0 + (t2 + j) * 16 + ln16;
        if (ncol < N) {
          const _Float16* bb = Bt + (size_t)ncol * K + k0;
          bfr[j] = ldfrag_h(bb + hf * 8, bb + 16 + hf * 8);
        } else {
          v16h bz = {};
          bfr[j] = bz;
        }
      }
#pragma unroll
      for (int j = 0; j < 2 && t2 + j < NT; ++j)
        acc[t2 + j] = __builtin_amdgcn_wmma_f32_16x16x32_f16(
            false, a, false, bfr[j], (short)0, acc[t2 + j], false, false);
    }
  }

#pragma unroll
  for (int t = 0; t < NT; ++t) {
    const int ncol = n0 + t * 16 + ln16;
    if (ncol >= N) continue;
    const float bv = bias ? bias[ncol] : 0.0f;
#pragma unroll
    for (int r = 0; r < 8; ++r) {
      const int row = m0 + hf * 8 + r;
      float vv = acc[t][r] + bv;
      if (HAS_RES) vv += res[(size_t)row * N + ncol];
      if (ACT == 1) vv = gelu_exact(vv);
      if (OUTM == 0) {
        ((float*)Outp)[(size_t)row * N + ncol] = vv;
      } else if (OUTM == 1) {
        ((_Float16*)Outp)[(size_t)row * N + ncol] = (_Float16)vv;
      } else {
        const int b  = row / SEQ;
        const int tt = row % SEQ;
        ((_Float16*)Outp)[((size_t)b * 512 + ncol) * SEQP + tt] = (_Float16)vv;
      }
    }
  }
}

// ---------------------------------------------------------------------------
// Flash attention, transposed orientation. One wave = one (b, h, q-tile).
// S^T = K @ Q^T (K-dim = head dim 64); O^T = V^T @ P^T (K-dim = 32 keys).
// qk: (B, SEQ, 1024) f16, col = j*512 + h*64 + d (j=0:Q, j=1:K)
// vt: (B, 512, SEQP) f16, row = h*64+d, pad tokens zeroed
// oatt: (B, SEQ, 512) f16, col = h*64 + d
// ---------------------------------------------------------------------------
__global__ void attn_flash(const void* __restrict__ qkp, const void* __restrict__ vtp,
                           const float* __restrict__ adj, const float* __restrict__ gam,
                           void* __restrict__ oattp) {
  const _Float16* qk   = (const _Float16*)qkp;
  const _Float16* vt   = (const _Float16*)vtp;
  _Float16*       oatt = (_Float16*)oattp;

  const int wave = threadIdx.x >> 5;
  const int lane = threadIdx.x & 31;
  const int hf   = lane >> 4;
  const int n    = lane & 15;                 // query within tile
  const int wid  = blockIdx.x * 8 + wave;
  if (wid >= BATCH * NHEAD * 33) return;      // wave-uniform
  const int b  = (wid / 33) >> 3;
  const int h  = (wid / 33) & 7;
  const int qt = wid % 33;
  const int q0 = qt * 16;
  const int qtok = q0 + n;
  const bool qv  = qtok < SEQ;
  const size_t base = (size_t)b * SEQ * QKW;

  // Q^T as B-operand fragments: contiguous chunks of the token row
  const int qc = imin(qtok, SEQ - 1);
  const _Float16* Qrow = qk + base + (size_t)qc * QKW + h * HDIM;
  const v16h bq0 = ldfrag_h(Qrow + hf * 16, Qrow + hf * 16 + 8);
  const v16h bq1 = ldfrag_h(Qrow + 32 + hf * 16, Qrow + 32 + hf * 16 + 8);

  float g0 = 0.0f, g1 = 0.0f;
  if (qv) {
    g0 = gam[((size_t)b * SEQ + qtok) * 2 + 0];
    g1 = gam[((size_t)b * SEQ + qtok) * 2 + 1];
  }

  // clamped adj row base for branchless gathers
  const float* adjrow = adj + ((size_t)b * 512 + imin(imax(qtok - 1, 0), 511)) * 512;

  const float scale = 0.125f;  // 64^-0.5
  float mrun = -1e30f, lrun = 0.0f;
  v8f zero = {};
  v8f oacc[4] = {zero, zero, zero, zero};

  for (int kt = 0; kt < NKEYBLK; ++kt) {
    // ---- scores S^T for 32 keys: per-tile K-fragment pair, 2 WMMAs ----
    v8f st[2];
#pragma unroll
    for (int t = 0; t < 2; ++t) {
      const int kc = imin(kt * 32 + t * 16 + n, SEQ - 1);
      const _Float16* Krow = qk + base + (size_t)kc * QKW + 512 + h * HDIM;
      const v16h ak0 = ldfrag_h(Krow + hf * 16, Krow + hf * 16 + 8);
      const v16h ak1 = ldfrag_h(Krow + 32 + hf * 16, Krow + 32 + hf * 16 + 8);
      v8f s = {};
      s = __builtin_amdgcn_wmma_f32_16x16x32_f16(false, ak0, false, bq0,
                                                 (short)0, s, false, false);
      s = __builtin_amdgcn_wmma_f32_16x16x32_f16(false, ak1, false, bq1,
                                                 (short)0, s, false, false);
      st[t] = s;
    }

    // ---- gate + adjacency (branchless clamped gather) + mask, block max ----
    float sc[16];
    float tmax = -1e30f;
#pragma unroll
    for (int i = 0; i < 16; ++i) {
      const int t = i >> 3, r = i & 7;
      const int key = kt * 32 + t * 16 + hf * 8 + r;
      const float al = adjrow[imin(imax(key - 1, 0), 511)];  // unconditional
      const bool inner = (qtok > 0) && qv && (key > 0) && (key < SEQ);
      const bool diag  = (qtok == 0) && (key == 0);
      const float av = inner ? al : (diag ? 1.0f : 0.0f);
      float sv = g0 * (st[t][r] * scale) + g1 * av;
      if (key >= SEQ) sv = -1e30f;
      sc[i] = sv;
      tmax = fmaxf(tmax, sv);
    }
    tmax = fmaxf(tmax, __shfl_xor(tmax, 16, 32));  // merge key halves
    const float mnew  = fmaxf(mrun, tmax);
    const float alpha = __expf(mrun - mnew);

    float ls = 0.0f;
#pragma unroll
    for (int i = 0; i < 16; ++i) {
      const float p = __expf(sc[i] - mnew);
      sc[i] = p;
      ls += p;
    }
    ls += __shfl_xor(ls, 16, 32);
    lrun = lrun * alpha + ls;
    mrun = mnew;
#pragma unroll
    for (int dt = 0; dt < 4; ++dt)
#pragma unroll
      for (int r = 0; r < 8; ++r) oacc[dt][r] *= alpha;

    // ---- build P^T B-operand via lane permutes (no LDS) ----
    v16h pv;
#pragma unroll
    for (int v = 0; v < 8; ++v) {
      const int r0 = (2 * v) & 7;
      const int r1 = (2 * v + 1) & 7;
      const int src = ((v >= 4) ? 16 : 0) + n;
      const float a0 = __shfl(sc[r0], src, 32);
      const float b0 = __shfl(sc[8 + r0], src, 32);
      const float a1 = __shfl(sc[r1], src, 32);
      const float b1 = __shfl(sc[8 + r1], src, 32);
      pv[2 * v]     = (_Float16)(hf ? b0 : a0);
      pv[2 * v + 1] = (_Float16)(hf ? b1 : a1);
    }

    // ---- O^T += V^T @ P^T: V fragments in pairs (16 transient VGPRs) ----
#pragma unroll
    for (int d2 = 0; d2 < 4; d2 += 2) {
      v16h avf[2];
#pragma unroll
      for (int j = 0; j < 2; ++j) {
        const _Float16* Vr =
            vt + ((size_t)b * 512 + h * HDIM + (d2 + j) * 16 + n) * SEQP + kt * 32;
        avf[j] = ldfrag_h(Vr + hf * 16, Vr + hf * 16 + 8);
      }
#pragma unroll
      for (int j = 0; j < 2; ++j)
        oacc[d2 + j] = __builtin_amdgcn_wmma_f32_16x16x32_f16(
            false, avf[j], false, pv, (short)0, oacc[d2 + j], false, false);
    }
  }

  // ---- epilogue: packed b128 stores, column n -> token q0+n ----
  if (qv) {
    const float inv = 1.0f / lrun;
    _Float16* orow = oatt + ((size_t)b * SEQ + qtok) * (NHEAD * HDIM) + h * HDIM;
#pragma unroll
    for (int dt = 0; dt < 4; ++dt) {
      union { v8h h; u32x4 u; } o;
#pragma unroll
      for (int r = 0; r < 8; ++r) o.h[r] = (_Float16)(oacc[dt][r] * inv);
      *(u32x4*)(orow + dt * 16 + hf * 8) = o.u;
    }
  }
}

// ---------------------------------------------------------------------------
// Weight prep: Wt(N x K) f16 = transpose(W(K x N) f32)
// ---------------------------------------------------------------------------
__global__ void wconv_t(const float* __restrict__ W, void* __restrict__ Wtp,
                        int K, int N) {
  const int idx = blockIdx.x * 256 + threadIdx.x;
  if (idx >= K * N) return;
  const int n = idx / K, k = idx % K;
  ((_Float16*)Wtp)[idx] = (_Float16)W[(size_t)k * N + n];
}

// zero V^T pad tokens [SEQ, SEQP) for all (b, h*64+d) rows
__global__ void vt_pad_zero(void* __restrict__ vtp) {
  const int idx = blockIdx.x * 256 + threadIdx.x;
  const int npad = SEQP - SEQ;  // 31
  if (idx >= BATCH * 512 * npad) return;
  const int row = idx / npad;
  const int t = SEQ + idx % npad;
  ((_Float16*)vtp)[(size_t)row * SEQP + t] = (_Float16)0.0f;
}

// ---------------------------------------------------------------------------
// Embedding: x = relu(nf@We1+be1)@We2 + be2 + (lapl@Wp + bp); cls token.
// ---------------------------------------------------------------------------
__global__ void embed_kernel(const float* __restrict__ nf, const float* __restrict__ lapl,
                             const float* __restrict__ cls, const float* __restrict__ clsp,
                             const float* __restrict__ Wp, const float* __restrict__ bp,
                             const float* __restrict__ We1, const float* __restrict__ be1,
                             const float* __restrict__ We2, const float* __restrict__ be2,
                             float* __restrict__ X) {
  const int tok = blockIdx.x;
  const int b = tok / SEQ, nn = tok % SEQ;
  const int d = threadIdx.x;
  __shared__ float e1[128];
  float out;
  if (nn == 0) {
    out = cls[d] + clsp[d];
  } else {
    const float* f = nf + ((size_t)b * 512 + (nn - 1)) * 16;
    for (int j = d; j < 128; j += 64) {
      float s = be1[j];
#pragma unroll
      for (int t = 0; t < 16; ++t) s += f[t] * We1[t * 128 + j];
      e1[j] = fmaxf(s, 0.0f);
    }
    __syncthreads();
    float s = be2[d];
    for (int j = 0; j < 128; ++j) s += e1[j] * We2[j * 64 + d];
    const float* lp = lapl + ((size_t)b * 512 + (nn - 1)) * 32;
    float p = bp[d];
#pragma unroll
    for (int t = 0; t < 32; ++t) p += lp[t] * Wp[t * 64 + d];
    out = s + p;
  }
  X[(size_t)tok * 64 + d] = out;
}

// ---------------------------------------------------------------------------
// LayerNorm (blockDim = D = 64|128); optional f16 copy of the output.
// ---------------------------------------------------------------------------
__global__ void ln_kernel(const float* __restrict__ in, const float* __restrict__ g,
                          const float* __restrict__ be, float* __restrict__ out,
                          void* __restrict__ outh, int D, int ldin) {
  const int row = blockIdx.x;
  const int d = threadIdx.x;
  __shared__ float red[128];
  const float v = in[(size_t)row * ldin + d];
  red[d] = v;
  __syncthreads();
  for (int s = blockDim.x >> 1; s > 0; s >>= 1) {
    if (d < s) red[d] += red[d + s];
    __syncthreads();
  }
  const float mu = red[0] / (float)D;
  __syncthreads();
  const float c = v - mu;
  red[d] = c * c;
  __syncthreads();
  for (int s = blockDim.x >> 1; s > 0; s >>= 1) {
    if (d < s) red[d] += red[d + s];
    __syncthreads();
  }
  const float var = red[0] / (float)D;
  const float r = c * rsqrtf(var + 1e-5f) * g[d] + be[d];
  out[(size_t)row * D + d] = r;
  if (outh) ((_Float16*)outh)[(size_t)row * D + d] = (_Float16)r;
}

// gam = exp(h @ Wg + bg), per token.
__global__ void gam_kernel(const float* __restrict__ h, const float* __restrict__ Wg,
                           const float* __restrict__ bg, float* __restrict__ gam) {
  const int tok = blockIdx.x * 256 + threadIdx.x;
  if (tok >= NTOK) return;
  const float* hr = h + (size_t)tok * 64;
  float g0 = bg[0], g1 = bg[1];
#pragma unroll
  for (int c = 0; c < 64; ++c) {
    const float hv = hr[c];
    g0 += hv * Wg[c * 2];
    g1 += hv * Wg[c * 2 + 1];
  }
  gam[tok * 2]     = __expf(g0);
  gam[tok * 2 + 1] = __expf(g1);
}

// ---------------------------------------------------------------------------
// Host launch
// ---------------------------------------------------------------------------
extern "C" void kernel_launch(void* const* d_in, const int* in_sizes, int n_in,
                              void* d_out, int out_size, void* d_ws, size_t ws_size,
                              hipStream_t stream) {
  (void)in_sizes; (void)n_in; (void)out_size; (void)ws_size;
  const float* node_feat = (const float*)d_in[0];
  const float* adj       = (const float*)d_in[1];
  const float* lapl      = (const float*)d_in[2];
  const float* cls  = (const float*)d_in[3];
  const float* clsp = (const float*)d_in[4];
  const float* Wp   = (const float*)d_in[5];
  const float* bp   = (const float*)d_in[6];
  const float* We1  = (const float*)d_in[7];
  const float* be1  = (const float*)d_in[8];
  const float* We2  = (const float*)d_in[9];
  const float* be2  = (const float*)d_in[10];
  const float* n1g  = (const float*)d_in[11];
  const float* n1b  = (const float*)d_in[12];
  const float* n2g  = (const float*)d_in[13];
  const float* n2b  = (const float*)d_in[14];
  const float* Wqkv = (const float*)d_in[15];
  const float* Wo   = (const float*)d_in[16];
  const float* bo   = (const float*)d_in[17];
  const float* Wg   = (const float*)d_in[18];
  const float* bg   = (const float*)d_in[19];
  const float* Wm1  = (const float*)d_in[20];
  const float* bm1  = (const float*)d_in[21];
  const float* Wm2  = (const float*)d_in[22];
  const float* bm2  = (const float*)d_in[23];
  const float* hg   = (const float*)d_in[24];
  const float* hb   = (const float*)d_in[25];
  const float* Wh   = (const float*)d_in[26];
  const float* bh   = (const float*)d_in[27];
  const float* Wp1  = (const float*)d_in[28];
  const float* bp1  = (const float*)d_in[29];
  const float* Wp2  = (const float*)d_in[30];
  const float* bp2  = (const float*)d_in[31];
  const float* Wp3  = (const float*)d_in[32];
  const float* bp3  = (const float*)d_in[33];
  const float* pg   = (const float*)d_in[34];
  const float* pb   = (const float*)d_in[35];
  const float* Wp4  = (const float*)d_in[36];
  const float* bp4  = (const float*)d_in[37];

  size_t off = 0;
  auto carve = [&](size_t bytes) -> void* {
    void* p = (char*)d_ws + off;
    off += (bytes + 255) & ~(size_t)255;
    return p;
  };
  float* X    = (float*)carve((size_t)NTOK * 64 * 4);
  float* H1   = (float*)carve((size_t)NTOK * 64 * 4);
  float* H2   = (float*)carve((size_t)NTOK * 64 * 4);
  float* GAM  = (float*)carve((size_t)NTOK * 2 * 4);
  float* CN   = (float*)carve((size_t)32 * 64 * 4);
  float* Z1   = (float*)carve((size_t)32 * 128 * 4);
  float* Z2   = (float*)carve((size_t)32 * 128 * 4);
  float* ZN   = (float*)carve((size_t)32 * 128 * 4);
  void* H1h   = carve((size_t)NTOK * 64 * 2);
  void* H2h   = carve((size_t)NTOK * 64 * 2);
  void* M1h   = carve((size_t)NTOK * 128 * 2);
  void* QK    = carve((size_t)NTOK * QKW * 2);
  void* VT    = carve((size_t)BATCH * 512 * SEQP * 2);
  void* OATT  = carve((size_t)NTOK * 512 * 2);
  void* WqkvT = carve((size_t)5 * 1536 * 64 * 2);
  void* WoT   = carve((size_t)5 * 64 * 512 * 2);
  void* Wm1T  = carve((size_t)5 * 128 * 64 * 2);
  void* Wm2T  = carve((size_t)5 * 64 * 128 * 2);
  void* WhT   = carve((size_t)64 * 64 * 2);
  void* Wp1T  = carve((size_t)128 * 64 * 2);
  void* Wp2T  = carve((size_t)128 * 128 * 2);
  void* Wp3T  = carve((size_t)128 * 128 * 2);
  void* Wp4T  = carve((size_t)1000 * 128 * 2);

  const dim3 blk(256);
  const int MT = NTOK / 16;  // 1026
  auto blocks = [](size_t n) { return (unsigned)((n + 255) / 256); };

  // ---- weight prep + V^T pad ----
  for (int i = 0; i < 5; ++i) {
    wconv_t<<<blocks(64 * 1536), blk, 0, stream>>>(
        Wqkv + (size_t)i * 64 * 1536, (char*)WqkvT + (size_t)i * 64 * 1536 * 2, 64, 1536);
    wconv_t<<<blocks(512 * 64), blk, 0, stream>>>(
        Wo + (size_t)i * 512 * 64, (char*)WoT + (size_t)i * 512 * 64 * 2, 512, 64);
    wconv_t<<<blocks(64 * 128), blk, 0, stream>>>(
        Wm1 + (size_t)i * 64 * 128, (char*)Wm1T + (size_t)i * 64 * 128 * 2, 64, 128);
    wconv_t<<<blocks(128 * 64), blk, 0, stream>>>(
        Wm2 + (size_t)i * 128 * 64, (char*)Wm2T + (size_t)i * 128 * 64 * 2, 128, 64);
  }
  wconv_t<<<blocks(64 * 64), blk, 0, stream>>>(Wh, WhT, 64, 64);
  wconv_t<<<blocks(64 * 128), blk, 0, stream>>>(Wp1, Wp1T, 64, 128);
  wconv_t<<<blocks(128 * 128), blk, 0, stream>>>(Wp2, Wp2T, 128, 128);
  wconv_t<<<blocks(128 * 128), blk, 0, stream>>>(Wp3, Wp3T, 128, 128);
  wconv_t<<<blocks(128 * 1000), blk, 0, stream>>>(Wp4, Wp4T, 128, 1000);
  vt_pad_zero<<<blocks((size_t)BATCH * 512 * (SEQP - SEQ)), blk, 0, stream>>>(VT);

  embed_kernel<<<NTOK, 64, 0, stream>>>(node_feat, lapl, cls, clsp, Wp, bp,
                                        We1, be1, We2, be2, X);

  for (int i = 0; i < 5; ++i) {
    const _Float16* WqkvTi = (const _Float16*)WqkvT + (size_t)i * 64 * 1536;
    ln_kernel<<<NTOK, 64, 0, stream>>>(X, n1g + i * 64, n1b + i * 64, H1, H1h, 64, 64);
    gam_kernel<<<blocks(NTOK), blk, 0, stream>>>(H1, Wg + i * 128, bg + i * 2, GAM);
    // Q|K = h @ Wqkv[:, :1024]  -> f16 row-major (NT=4)
    gemm16<true, 0, false, 1, 4><<<dim3(2, MT), blk, 0, stream>>>(
        H1h, WqkvTi, nullptr, nullptr, QK, NTOK, 1024, 64);
    // V = h @ Wqkv[:, 1024:]    -> f16 transposed V^T layout (NT=4)
    gemm16<true, 0, false, 2, 4><<<dim3(1, MT), blk, 0, stream>>>(
        H1h, WqkvTi + (size_t)1024 * 64, nullptr, nullptr, VT, NTOK, 512, 64);
    attn_flash<<<(BATCH * NHEAD * 33) / 8, blk, 0, stream>>>(QK, VT, adj, GAM, OATT);
    // x = oatt @ Wo + bo + h
    gemm16<true, 0, true, 0, 1><<<dim3(1, MT), blk, 0, stream>>>(
        OATT, (const _Float16*)WoT + (size_t)i * 64 * 512, bo + i * 64, H1, X,
        NTOK, 64, 512);
    ln_kernel<<<NTOK, 64, 0, stream>>>(X, n2g + i * 64, n2b + i * 64, H2, H2h, 64, 64);
    // m1 = gelu(h2 @ Wm1 + bm1) -> f16
    gemm16<true, 1, false, 1, 1><<<dim3(1, MT), blk, 0, stream>>>(
        H2h, (const _Float16*)Wm1T + (size_t)i * 128 * 64, bm1 + i * 128, nullptr,
        M1h, NTOK, 128, 64);
    // x = m1 @ Wm2 + bm2 + h2
    gemm16<true, 0, true, 0, 1><<<dim3(1, MT), blk, 0, stream>>>(
        M1h, (const _Float16*)Wm2T + (size_t)i * 64 * 128, bm2 + i * 64, H2, X,
        NTOK, 64, 128);
  }

  // ---- head ----
  float* dout = (float*)d_out;
  ln_kernel<<<32, 64, 0, stream>>>(X, hg, hb, CN, nullptr, 64, SEQ * 64);
  gemm16<false, 0, false, 0, 1><<<dim3(1, 2), blk, 0, stream>>>(
      CN, WhT, bh, nullptr, dout, 32, 64, 64);                   // c -> d_out[0:2048)
  gemm16<false, 1, false, 0, 1><<<dim3(1, 2), blk, 0, stream>>>(
      dout, Wp1T, bp1, nullptr, Z1, 32, 128, 64);
  gemm16<false, 1, false, 0, 1><<<dim3(1, 2), blk, 0, stream>>>(
      Z1, Wp2T, bp2, nullptr, Z2, 32, 128, 128);
  gemm16<false, 1, false, 0, 1><<<dim3(1, 2), blk, 0, stream>>>(
      Z2, Wp3T, bp3, nullptr, Z1, 32, 128, 128);
  ln_kernel<<<32, 128, 0, stream>>>(Z1, pg, pb, ZN, nullptr, 128, 128);
  gemm16<false, 0, false, 0, 1><<<dim3(8, 2), blk, 0, stream>>>(
      ZN, Wp4T, bp4, nullptr, dout + 2048, 32, 1000, 128);       // z -> d_out[2048:)
}
